// DFA_34497177322000
// MI455X (gfx1250) — matrix operand order repaired and checked by reading, not compiled, not run on hardware
//
#include <hip/hip_runtime.h>

#define SEQLEN 2048
#define BATCH  64
#define EDIM   128
#define HDIM   256
#define CLS    16
#define NWG    8
#define NTHR   256
#define XS     136   // padded LDS stride for x tile (bank-spread for b128 loads)
#define HS     264   // padded LDS stride for h tile

typedef __bf16 v8bf  __attribute__((ext_vector_type(8)));
typedef __bf16 v16bf __attribute__((ext_vector_type(16)));
typedef float  v8f   __attribute__((ext_vector_type(8)));

union V16 { v16bf v; v8bf h[2]; };

__device__ __forceinline__ v8f wmma_bf16(v16bf a, v16bf b, v8f c) {
  // (neg_a, A, neg_b, B, c_mod, C, reuse_a, reuse_b)
  return __builtin_amdgcn_wmma_f32_16x16x32_bf16(false, a, false, b, (short)0, c,
                                                 false, false);
}

// A fragment (16x32 bf16) from LDS, row-major with stride `st` elements.
// ISA layout: lane(<16) holds M=m0+lane, K = [k0+hi*8 .. +7] and [k0+16+hi*8 .. +7]
__device__ __forceinline__ v16bf load_a(const __bf16* S, int st, int m0, int k0,
                                        int laneLo, int hi) {
  const __bf16* p = S + (m0 + laneLo) * st + k0 + hi * 8;
  V16 u;
  u.h[0] = *(const v8bf*)p;
  u.h[1] = *(const v8bf*)(p + 16);
  return u.v;
}

// B fragment (32x16 bf16) from global bf16 weights W[j][k] (row-major, stride `st`).
// B[k][n] = W[n][k]; lane holds N=j0+lane%16, K = [k0+hi*8 .. +7] and [k0+16+hi*8 .. +7].
__device__ __forceinline__ v16bf load_b(const __bf16* __restrict__ W, int st, int j0,
                                        int k0, int laneLo, int hi) {
  const __bf16* p = W + (j0 + laneLo) * st + k0 + hi * 8;
  V16 u;
  u.h[0] = *(const v8bf*)p;
  u.h[1] = *(const v8bf*)(p + 16);
  return u.v;
}

__global__ void gru_prep(const float* __restrict__ wih_f, const float* __restrict__ whh_f,
                         __bf16* __restrict__ wih, __bf16* __restrict__ whh,
                         __bf16* __restrict__ hbuf, unsigned* __restrict__ cnt) {
  int tid = blockIdx.x * blockDim.x + threadIdx.x;
  int str = gridDim.x * blockDim.x;
  if (tid == 0) *cnt = 0u;
  for (int i = tid; i < 3 * HDIM * EDIM; i += str) wih[i] = (__bf16)wih_f[i];
  for (int i = tid; i < 3 * HDIM * HDIM; i += str) whh[i] = (__bf16)whh_f[i];
  for (int i = tid; i < 2 * BATCH * HDIM; i += str) hbuf[i] = (__bf16)0.0f;
}

__global__ __launch_bounds__(NTHR)
void gru_persistent(const int* __restrict__ seq, const float* __restrict__ emb,
                    const float* __restrict__ b_ih, const float* __restrict__ b_hh,
                    const float* __restrict__ W_out, const float* __restrict__ b_out,
                    const __bf16* __restrict__ wih, const __bf16* __restrict__ whh,
                    __bf16* __restrict__ hbuf, unsigned* __restrict__ cnt,
                    float* __restrict__ out) {
  __shared__ __bf16 sX[BATCH * XS];   // x tile, bf16
  __shared__ __bf16 sH[BATCH * HS];   // h tile, bf16
  __shared__ float  sc[BATCH * CLS];  // classifier scratch

  const int tid    = threadIdx.x;
  const int wg     = blockIdx.x;
  const int wave   = tid >> 5;
  const int lane   = tid & 31;
  const int laneLo = lane & 15;
  const int hi     = lane >> 4;
  const int mt     = wave & 3;          // M tile (rows 16*mt..)
  const int nt     = wave >> 2;         // N tile within this WG's 32-col slice
  const int colh0  = wg * 32 + nt * 16; // base h-column
  const int colh   = colh0 + laneLo;

  // step-invariant biases for this lane's output column
  const float br  = b_ih[colh] + b_hh[colh];
  const float bz  = b_ih[HDIM + colh] + b_hh[HDIM + colh];
  const float bin = b_ih[2 * HDIM + colh];
  const float bhn = b_hh[2 * HDIM + colh];

  float* __restrict__ hiddens = out + BATCH * CLS;

  for (int s = 0; s < SEQLEN; ++s) {
    const __bf16* __restrict__ hr = hbuf + (s & 1) * (BATCH * HDIM);
    __bf16* __restrict__       hw = hbuf + ((s + 1) & 1) * (BATCH * HDIM);

    // gather x = emb[seq[s,:]] -> bf16 LDS
    for (int i = tid * 4; i < BATCH * EDIM; i += NTHR * 4) {
      int row = i >> 7, col = i & 127;
      int v = seq[s * BATCH + row];
      const float4 f = *(const float4*)(emb + (long)v * EDIM + col);
      __bf16* d = sX + row * XS + col;
      d[0] = (__bf16)f.x; d[1] = (__bf16)f.y; d[2] = (__bf16)f.z; d[3] = (__bf16)f.w;
    }
    // load full h state (bf16) -> LDS
    for (int i = tid * 8; i < BATCH * HDIM; i += NTHR * 8) {
      int row = i >> 8, col = i & 255;
      *(v8bf*)(sH + row * HS + col) = *(const v8bf*)(hr + i);
    }
    __syncthreads();

    v8f ar = {}, az = {}, ain = {}, ahn = {};
    const int jr = colh0, jz = HDIM + colh0, jn = 2 * HDIM + colh0;
#pragma unroll
    for (int k0 = 0; k0 < EDIM; k0 += 32) {     // x @ W_ih^T  (r, z, i_n)
      v16bf a = load_a(sX, XS, mt * 16, k0, laneLo, hi);
      ar  = wmma_bf16(a, load_b(wih, EDIM, jr, k0, laneLo, hi), ar);
      az  = wmma_bf16(a, load_b(wih, EDIM, jz, k0, laneLo, hi), az);
      ain = wmma_bf16(a, load_b(wih, EDIM, jn, k0, laneLo, hi), ain);
    }
#pragma unroll
    for (int k0 = 0; k0 < HDIM; k0 += 32) {     // h @ W_hh^T  (r, z, h_n)
      v16bf a = load_a(sH, HS, mt * 16, k0, laneLo, hi);
      ar  = wmma_bf16(a, load_b(whh, HDIM, jr, k0, laneLo, hi), ar);
      az  = wmma_bf16(a, load_b(whh, HDIM, jz, k0, laneLo, hi), az);
      ahn = wmma_bf16(a, load_b(whh, HDIM, jn, k0, laneLo, hi), ahn);
    }

    // epilogue; C/D layout: VGPR g, lane -> (M = g + 8*hi, N = lane%16)
#pragma unroll
    for (int g = 0; g < 8; ++g) {
      int m = mt * 16 + g + 8 * hi;
      float r  = 1.0f / (1.0f + __expf(-(ar[g] + br)));
      float z  = 1.0f / (1.0f + __expf(-(az[g] + bz)));
      float n  = tanhf(ain[g] + bin + r * (ahn[g] + bhn));
      float ho = (float)sH[m * HS + colh];
      float hn2 = (1.0f - z) * n + z * ho;
      hiddens[(long)s * (BATCH * HDIM) + m * HDIM + colh] = hn2;
      hw[m * HDIM + colh] = (__bf16)hn2;
    }

    // 8-WG grid barrier (monotonic counter; cnt zeroed by prep each launch)
    __syncthreads();
    if (tid == 0) {
      __threadfence();
      atomicAdd(cnt, 1u);
      unsigned target = (unsigned)(s + 1) * NWG;
      while (__hip_atomic_load(cnt, __ATOMIC_ACQUIRE, __HIP_MEMORY_SCOPE_AGENT) < target)
        __builtin_amdgcn_s_sleep(1);
    }
    __syncthreads();
  }

  // ---- classifier + log_softmax on WG0 (final h lives in hbuf[0]) ----
  if (wg != 0) return;
  const __bf16* __restrict__ hf = hbuf;
  for (int i = tid; i < BATCH * CLS; i += NTHR) {
    int b = i >> 4, c = i & 15;
    float acc = b_out[c];
    for (int k = 0; k < HDIM; ++k)
      acc += (float)hf[b * HDIM + k] * W_out[c * HDIM + k];
    sc[i] = acc;
  }
  __syncthreads();
  if (tid < BATCH) {
    float m = -1e30f;
    for (int c = 0; c < CLS; ++c) m = fmaxf(m, sc[tid * CLS + c]);
    float sum = 0.0f;
    for (int c = 0; c < CLS; ++c) sum += __expf(sc[tid * CLS + c] - m);
    float lse = m + logf(sum);
    for (int c = 0; c < CLS; ++c) out[tid * CLS + c] = sc[tid * CLS + c] - lse;
  }
}

extern "C" void kernel_launch(void* const* d_in, const int* in_sizes, int n_in,
                              void* d_out, int out_size, void* d_ws, size_t ws_size,
                              hipStream_t stream) {
  const int*   seq   = (const int*)  d_in[0];
  const float* emb   = (const float*)d_in[1];
  const float* W_ih  = (const float*)d_in[2];
  const float* W_hh  = (const float*)d_in[3];
  const float* b_ih  = (const float*)d_in[4];
  const float* b_hh  = (const float*)d_in[5];
  const float* W_out = (const float*)d_in[6];
  const float* b_out = (const float*)d_in[7];

  // workspace carve (~656 KB total)
  char* ws = (char*)d_ws;
  __bf16*   wih  = (__bf16*)(ws);             // 768*128 bf16 = 196608 B
  __bf16*   whh  = (__bf16*)(ws + 196608);    // 768*256 bf16 = 393216 B
  __bf16*   hbuf = (__bf16*)(ws + 589824);    // 2*64*256 bf16 = 65536 B
  unsigned* cnt  = (unsigned*)(ws + 655360);  // barrier counter

  gru_prep<<<256, 256, 0, stream>>>(W_ih, W_hh, wih, whh, hbuf, cnt);
  gru_persistent<<<NWG, NTHR, 0, stream>>>(seq, emb, b_ih, b_hh, W_out, b_out,
                                           wih, whh, hbuf, cnt, (float*)d_out);
}
